// BasicGNN_60765197304392
// MI455X (gfx1250) — matrix operand order repaired and checked by reading, not compiled, not run on hardware
//
#include <hip/hip_runtime.h>

// ---------------------------------------------------------------------------
// 2-layer GCN for MI455X (gfx1250, wave32, WGP with 320KB LDS).
//  - conv1 feature transform X@W1 on the bf16 WMMA pipe
//    (v_wmma_f32_16x16x32_bf16, f32 accumulate).
//  - W1^T (64KB bf16) staged in LDS once per block; 8 waves/block each own
//    one 16-row M tile and sweep all 8 N tiles -> A loaded once per K step,
//    64 WMMAs per wave, ds_load-fed B operands.
//  - edge aggregation: gather + f32 atomics; xw (51MB) and agg (51MB) both
//    fit in the 192MB L2, so the random scatter stays on-chip.
// ---------------------------------------------------------------------------

typedef __attribute__((ext_vector_type(16))) __bf16 v16bf;
typedef __attribute__((ext_vector_type(8)))  float  v8f;

#define D_IN  256
#define H_MID 128

// ---------------- degree / normalization ----------------
__global__ void k_init_deg(float* __restrict__ deg, int N) {
  int i = blockIdx.x * blockDim.x + threadIdx.x;
  if (i < N) deg[i] = 1.0f;                 // self-loop contributes 1
}

__global__ void k_deg_edges(const long long* __restrict__ dst,
                            float* __restrict__ deg, long long E) {
  long long e = (long long)blockIdx.x * blockDim.x + threadIdx.x;
  if (e < E) atomicAdd(&deg[(int)dst[e]], 1.0f);
}

__global__ void k_dinv(float* __restrict__ deg, int N) {
  int i = blockIdx.x * blockDim.x + threadIdx.x;
  if (i < N) deg[i] = rsqrtf(deg[i]);       // deg >= 1 always (self-loop)
}

// ---------------- bf16 conversion ----------------
__global__ void k_cvt_x(const float* __restrict__ x, __bf16* __restrict__ xb,
                        long long n) {
  long long i = (long long)blockIdx.x * blockDim.x + threadIdx.x;
  if (i < n) xb[i] = (__bf16)x[i];
}

// W1 [K=256][N=128] row-major -> Wt bf16 [N=128][K=256] (column-major W1),
// so each B-fragment lane reads 16 contiguous bf16.
__global__ void k_cvt_w1t(const float* __restrict__ w1, __bf16* __restrict__ wt) {
  int i = blockIdx.x * blockDim.x + threadIdx.x;
  if (i < D_IN * H_MID) {
    int k = i / H_MID, n = i % H_MID;
    wt[n * D_IN + k] = (__bf16)w1[i];
  }
}

// ---------------- WMMA GEMM: xw[N,128] = xb[N,256] @ W1 ----------------
// Block = 256 threads = 8 waves; block covers 128 rows (8 M tiles).
// Whole W1^T staged in LDS (exactly 64KB) once per block.
// Fragment layouts used here (A/B K order contiguous, not the ISA table; C/D per ISA 7.12.2), lane = (half<<4)|r:
//   A (16-bit 16x32): elems 0..15 -> K = half*16 + j, row = r
//   B: lanes 0-15 hold K=0..15, lanes 16-31 K=16..31, col = r
//   C/D: VGPR v -> row = half*8 + v, col = r
__global__ void __launch_bounds__(256)
k_gemm_bf16(const __bf16* __restrict__ xb, const __bf16* __restrict__ wt,
            float* __restrict__ xw, int N) {
  __shared__ __bf16 lds_wt[D_IN * H_MID];   // 65536 bytes

  // Stage W1^T into LDS: 4096 x 16B, 16 per thread.
  {
    const uint4* gsrc = (const uint4*)wt;
    uint4*       ldst = (uint4*)lds_wt;
    #pragma unroll
    for (int i = 0; i < (D_IN * H_MID * 2) / 16 / 256; ++i)
      ldst[threadIdx.x + i * 256] = gsrc[threadIdx.x + i * 256];
  }
  __syncthreads();

  const int lane = threadIdx.x & 31;
  const int wv   = threadIdx.x >> 5;        // wave id -> M tile within block
  const int r    = lane & 15;
  const int half = lane >> 4;
  const int m0   = (blockIdx.x * 8 + wv) * 16;
  if (m0 >= N) return;

  int arow_idx = m0 + r;
  if (arow_idx >= N) arow_idx = N - 1;      // pad guard (N%16==0 in practice)
  const __bf16* arow = xb + (size_t)arow_idx * D_IN;

  v8f acc[8];
  #pragma unroll
  for (int n = 0; n < 8; ++n) acc[n] = (v8f){};

  #pragma unroll
  for (int kb = 0; kb < D_IN; kb += 32) {
    // A fragment: loaded ONCE per K step, reused across all 8 N tiles.
    v16bf a;
    #pragma unroll
    for (int j = 0; j < 8; ++j) {
      a[j]     = arow[kb + half * 16 + j];
      a[j + 8] = arow[kb + half * 16 + 8 + j];
    }
    #pragma unroll
    for (int n = 0; n < 8; ++n) {
      const __bf16* bbase = lds_wt + (n * 16 + r) * D_IN + kb + half * 16;
      v16bf b;
      #pragma unroll
      for (int j = 0; j < 16; ++j) b[j] = bbase[j];
      acc[n] = __builtin_amdgcn_wmma_f32_16x16x32_bf16(
          /*neg_a=*/false, a, /*neg_b=*/false, b,
          /*c_mod=*/(short)0, acc[n], /*reuse_a=*/false, /*reuse_b=*/false);
    }
  }

  #pragma unroll
  for (int n = 0; n < 8; ++n) {
    #pragma unroll
    for (int v = 0; v < 8; ++v) {
      int row = m0 + half * 8 + v;
      if (row < N) xw[(size_t)row * H_MID + n * 16 + r] = acc[n][v];
    }
  }
}

// ---------------- conv1 aggregation ----------------
// Self-loop init: agg[i] = dinv[i]^2 * xw[i] (also fully initializes the
// poisoned workspace region).
__global__ void k_selfloop1(const float* __restrict__ xw,
                            const float* __restrict__ dinv,
                            float* __restrict__ agg, long long total) {
  long long i = (long long)blockIdx.x * blockDim.x + threadIdx.x;
  if (i < total) {
    float di = dinv[(int)(i >> 7)];        // /128
    agg[i] = di * di * xw[i];
  }
}

// One wave per edge: 32 lanes x float4 = 128 features.
__global__ void k_scatter1(const long long* __restrict__ src,
                           const long long* __restrict__ dst,
                           const float* __restrict__ dinv,
                           const float* __restrict__ xw,
                           float* __restrict__ agg, long long E) {
  long long t = (long long)blockIdx.x * blockDim.x + threadIdx.x;
  long long e = t >> 5;
  int lane = (int)(t & 31);
  if (e >= E) return;
  int s = (int)src[e], d = (int)dst[e];
  float norm = dinv[s] * dinv[d];
  const float4* xr = (const float4*)(xw + (size_t)s * H_MID);
  float4 v = xr[lane];                      // global_load_b128, L2-resident
  float* od = agg + (size_t)d * H_MID + lane * 4;
  atomicAdd(od + 0, norm * v.x);
  atomicAdd(od + 1, norm * v.y);
  atomicAdd(od + 2, norm * v.z);
  atomicAdd(od + 3, norm * v.w);
}

__global__ void k_relu_bias(float* __restrict__ agg,
                            const float* __restrict__ b1, long long total) {
  long long i = (long long)blockIdx.x * blockDim.x + threadIdx.x;
  if (i < total) {
    float v = agg[i] + b1[(int)(i & (H_MID - 1))];
    agg[i] = v > 0.0f ? v : 0.0f;
  }
}

// ---------------- conv2 ----------------
// z[i] = dot(h[i,:], W2[:,0]); one wave per node, shuffle reduction.
__global__ void k_dot_w2(const float* __restrict__ h,
                         const float* __restrict__ w2,
                         float* __restrict__ z, int N) {
  int t = blockIdx.x * blockDim.x + threadIdx.x;
  int node = t >> 5;
  int lane = t & 31;
  if (node >= N) return;
  const float* hr = h + (size_t)node * H_MID;
  float p = 0.0f;
  #pragma unroll
  for (int j = 0; j < 4; ++j)
    p += hr[lane + 32 * j] * w2[lane + 32 * j];
  #pragma unroll
  for (int off = 16; off > 0; off >>= 1)
    p += __shfl_xor(p, off, 32);
  if (lane == 0) z[node] = p;
}

// out[i] = dinv[i]^2 * z[i] + b2   (self-loop + bias; fully inits d_out)
__global__ void k_out_init(const float* __restrict__ z,
                           const float* __restrict__ dinv,
                           const float* __restrict__ b2,
                           float* __restrict__ out, int N) {
  int i = blockIdx.x * blockDim.x + threadIdx.x;
  if (i < N) {
    float di = dinv[i];
    out[i] = di * di * z[i] + b2[0];
  }
}

__global__ void k_scatter2(const long long* __restrict__ src,
                           const long long* __restrict__ dst,
                           const float* __restrict__ dinv,
                           const float* __restrict__ z,
                           float* __restrict__ out, long long E) {
  long long e = (long long)blockIdx.x * blockDim.x + threadIdx.x;
  if (e >= E) return;
  int s = (int)src[e], d = (int)dst[e];
  atomicAdd(&out[d], dinv[s] * dinv[d] * z[s]);
}

// ---------------------------------------------------------------------------
static inline unsigned blocks_for(long long n, int bs) {
  return (unsigned)((n + bs - 1) / bs);
}

extern "C" void kernel_launch(void* const* d_in, const int* in_sizes, int n_in,
                              void* d_out, int out_size, void* d_ws, size_t ws_size,
                              hipStream_t stream) {
  const float*     x    = (const float*)d_in[0];       // [N,256]
  const long long* eidx = (const long long*)d_in[1];   // [2,E] int64
  const float*     W1   = (const float*)d_in[2];       // [256,128]
  const float*     b1   = (const float*)d_in[3];       // [128]
  const float*     W2   = (const float*)d_in[4];       // [128,1]
  const float*     b2   = (const float*)d_in[5];       // [1]
  float*           out  = (float*)d_out;               // [N]

  const long long N = (long long)in_sizes[0] / D_IN;
  const long long E = (long long)in_sizes[1] / 2;
  const long long* src = eidx;
  const long long* dst = eidx + E;

  // Workspace layout (~104 MB for N=100K):
  //   [0, N*512)          : xb   bf16 [N,256]   -- aliased as agg1 f32 [N,128]
  //   [N*512, N*1024)     : xw   f32  [N,128]
  //   then deg/dinv f32[N], z f32[N], wt bf16[128*256]
  char* w = (char*)d_ws;
  __bf16* xb   = (__bf16*)w;
  float*  agg1 = (float*)w;                 // alias: xb dead after GEMM
  float*  xw   = (float*)(w + (size_t)N * 512);
  char*   p    = w + (size_t)N * 1024;
  float*  deg  = (float*)p;  p += (size_t)N * 4;   // rsqrt'd in place -> dinv
  float*  z    = (float*)p;  p += (size_t)N * 4;
  __bf16* wt   = (__bf16*)p;

  const int BS = 256;

  // 1) symmetric normalization: deg -> dinv
  k_init_deg <<<blocks_for(N, BS), BS, 0, stream>>>(deg, (int)N);
  k_deg_edges<<<blocks_for(E, BS), BS, 0, stream>>>(dst, deg, E);
  k_dinv     <<<blocks_for(N, BS), BS, 0, stream>>>(deg, (int)N);
  float* dinv = deg;

  // 2) bf16 operands for WMMA
  k_cvt_x  <<<blocks_for(N * D_IN, BS), BS, 0, stream>>>(x, xb, N * D_IN);
  k_cvt_w1t<<<blocks_for((long long)D_IN * H_MID, BS), BS, 0, stream>>>(W1, wt);

  // 3) xw = X @ W1 on the matrix pipe.
  //    8 waves/block, each owns one M tile and all 8 N tiles; block = 128 rows.
  k_gemm_bf16<<<blocks_for(N, 128), 256, 0, stream>>>(xb, wt, xw, (int)N);

  // 4) conv1 aggregation (self-loop init, then edge scatter), relu + bias
  k_selfloop1<<<blocks_for(N * H_MID, BS), BS, 0, stream>>>(xw, dinv, agg1, N * H_MID);
  k_scatter1 <<<blocks_for(E * 32, BS), BS, 0, stream>>>(src, dst, dinv, xw, agg1, E);
  k_relu_bias<<<blocks_for(N * H_MID, BS), BS, 0, stream>>>(agg1, b1, N * H_MID);

  // 5) conv2: per-node dot with W2, then scalar aggregation into d_out
  k_dot_w2  <<<blocks_for(N * 32, BS), BS, 0, stream>>>(agg1, W2, z, (int)N);
  k_out_init<<<blocks_for(N, BS), BS, 0, stream>>>(z, dinv, b2, out, (int)N);
  k_scatter2<<<blocks_for(E, BS), BS, 0, stream>>>(src, dst, dinv, z, out, E);
}
